// _RBFKANLayer_24644522344533
// MI455X (gfx1250) — matrix operand order repaired and checked by reading, not compiled, not run on hardware
//
#include <hip/hip_runtime.h>

typedef __attribute__((ext_vector_type(16))) _Float16 v16h;
typedef __attribute__((ext_vector_type(8)))  float    v8f;

#define TILE_M   128
#define TILE_N   128
#define IC       8          // inputs per spline K-chunk
#define GRIDSZ   12
#define KC       (IC*GRIDSZ)  // 96 K values per chunk = 3 wmma k-steps
#define KSTEPS   3
#define NTHREADS 256

// out[b,o] = sum_{i,g} exp(-0.5*((x[b,i]-c[g])/h)^2) * coeff[i,o,g]
//          + sum_i x[b,i]*base_w[o,i] + base_b[o]
__global__ __launch_bounds__(NTHREADS)
void rbfkan_wmma_kernel(const float* __restrict__ x,
                        const float* __restrict__ coeff,
                        const float* __restrict__ base_w,
                        const float* __restrict__ base_b,
                        const float* __restrict__ centers,
                        float* __restrict__ out)
{
    const int I = 512, O = 512, G = GRIDSZ;
    const float inv_h = 1.375f;  // 1 / (8/11)

    // Fragment-swizzled staging buffers (f16).
    // A: [8 mfrags][3 ksteps][32 lanes][16 halves]
    // B: [3 ksteps][8 nfrags][32 lanes(=N col + K bit 3)][16 halves(=K)]
    __shared__ __align__(32) _Float16 lds_a[8 * KSTEPS * 32 * 16];
    __shared__ __align__(32) _Float16 lds_b[KSTEPS * 8 * 32 * 16];
    __shared__ float xs[TILE_M * IC];
    __shared__ float cent_s[16];

    const int tid  = threadIdx.x;
    const int lane = tid & 31;
    const int wave = tid >> 5;
    const int wm   = wave & 1;   // 2 waves along M (64 rows each)
    const int wn   = wave >> 1;  // 4 waves along N (32 cols each)

    const int m0 = blockIdx.y * TILE_M;
    const int n0 = blockIdx.x * TILE_N;

    if (tid < G) cent_s[tid] = centers[tid];

    v8f acc[4][2];
    const v8f vzero = {0.f, 0.f, 0.f, 0.f, 0.f, 0.f, 0.f, 0.f};
#pragma unroll
    for (int mf = 0; mf < 4; ++mf)
#pragma unroll
        for (int nf = 0; nf < 2; ++nf)
            acc[mf][nf] = vzero;

    // ---------------- spline GEMM: K = I*G = 6144 ----------------
    for (int i0 = 0; i0 < I; i0 += IC) {
        // stage x tile (128 rows x 8 inputs)
        for (int idx = tid; idx < TILE_M * IC; idx += NTHREADS) {
            int r = idx >> 3, c = idx & 7;
            xs[idx] = x[(size_t)(m0 + r) * I + i0 + c];
        }
        __syncthreads();  // also separates previous chunk's WMMAs from LDS refill

        // A-tile: compute RBF basis, write in A-fragment order
        for (int idx = tid; idx < TILE_M * KC; idx += NTHREADS) {
            int r  = idx / KC;          // row 0..127
            int k  = idx - r * KC;      // 0..95
            int ii = k / G;
            int g  = k - ii * G;
            float d = (xs[r * IC + ii] - cent_s[g]) * inv_h;
            float v = __expf(-0.5f * d * d);
            int mf = r >> 4, m = r & 15;
            int ks = k >> 5, kk = k & 31;
            int lhi  = (kk & 8) << 1;                 // lane group (K bit 3)
            int half = (kk & 7) | ((kk >> 4) << 3);   // K bits 2:0 and 4
            lds_a[(((mf * KSTEPS + ks) * 32 + (m + lhi)) << 4) + half] = (_Float16)v;
        }
        // B-tile: coeff[i, n0+n, g] -> B-fragment order (lane = N col + K bit 3, half = K)
        for (int idx = tid; idx < KC * TILE_N; idx += NTHREADS) {
            int n  = idx / KC;          // 0..127
            int k  = idx - n * KC;      // 0..95
            int ii = k / G;
            int g  = k - ii * G;
            float v = coeff[((size_t)(i0 + ii) * O + (n0 + n)) * G + g];
            int ks = k >> 5, kk = k & 31;
            lds_b[(((ks * 8 + (n >> 4)) * 32 + ((n & 15) + ((kk & 8) << 1))) << 4) + ((kk & 7) | ((kk >> 4) << 3))] = (_Float16)v;
        }
        if (i0 + IC < I) {  // pull next coeff slice toward L2 (global_prefetch_b8)
            __builtin_prefetch(&coeff[((size_t)(i0 + IC + (tid >> 5)) * O +
                                       n0 + (tid & 31) * 4) * G], 0, 1);
        }
        __syncthreads();

#pragma unroll
        for (int ks = 0; ks < KSTEPS; ++ks) {
            v16h bf0 = *(const v16h*)&lds_b[((ks * 8 + (wn * 2 + 0)) * 32 + lane) << 4];
            v16h bf1 = *(const v16h*)&lds_b[((ks * 8 + (wn * 2 + 1)) * 32 + lane) << 4];
#pragma unroll
            for (int mf = 0; mf < 4; ++mf) {
                v16h af = *(const v16h*)&lds_a[(((wm * 4 + mf) * KSTEPS + ks) * 32 + lane) << 4];
                acc[mf][0] = __builtin_amdgcn_wmma_f32_16x16x32_f16(
                    false, af, false, bf0, (short)0, acc[mf][0], false, false);
                acc[mf][1] = __builtin_amdgcn_wmma_f32_16x16x32_f16(
                    false, af, false, bf1, (short)0, acc[mf][1], false, false);
            }
        }
    }

    // ---------------- residual GEMM: out += x @ base_w^T (K = 512) ----------------
    for (int i0 = 0; i0 < I; i0 += 32) {
        __syncthreads();  // previous WMMAs done before refill
        for (int idx = tid; idx < TILE_M * 32; idx += NTHREADS) {
            int r = idx >> 5, kk = idx & 31;
            float v = x[(size_t)(m0 + r) * I + i0 + kk];
            int mf = r >> 4, m = r & 15;
            int lhi  = (kk & 8) << 1;
            int half = (kk & 7) | ((kk >> 4) << 3);
            lds_a[((mf * KSTEPS * 32 + (m + lhi)) << 4) + half] = (_Float16)v;
        }
        for (int idx = tid; idx < 32 * TILE_N; idx += NTHREADS) {
            int n = idx >> 5, kk = idx & 31;
            float v = base_w[(size_t)(n0 + n) * I + i0 + kk];
            lds_b[(((n >> 4) * 32 + ((n & 15) + ((kk & 8) << 1))) << 4) + ((kk & 7) | ((kk >> 4) << 3))] = (_Float16)v;
        }
        __syncthreads();

        v16h bf0 = *(const v16h*)&lds_b[((wn * 2 + 0) * 32 + lane) << 4];
        v16h bf1 = *(const v16h*)&lds_b[((wn * 2 + 1) * 32 + lane) << 4];
#pragma unroll
        for (int mf = 0; mf < 4; ++mf) {
            v16h af = *(const v16h*)&lds_a[((wm * 4 + mf) * KSTEPS * 32 + lane) << 4];
            acc[mf][0] = __builtin_amdgcn_wmma_f32_16x16x32_f16(
                false, af, false, bf0, (short)0, acc[mf][0], false, false);
            acc[mf][1] = __builtin_amdgcn_wmma_f32_16x16x32_f16(
                false, af, false, bf1, (short)0, acc[mf][1], false, false);
        }
    }

    // ---------------- epilogue: + bias, store f32 ----------------
    // C/D layout: vgpr r, lanes 0-15 -> (M=r, N=lane); lanes 16-31 -> (M=8+r, N=lane-16)
    const int col  = lane & 15;
    const int rsel = (lane >> 4) << 3;
#pragma unroll
    for (int nf = 0; nf < 2; ++nf) {
        int ng = n0 + wn * 32 + nf * 16 + col;
        float bias = base_b[ng];
#pragma unroll
        for (int mf = 0; mf < 4; ++mf) {
#pragma unroll
            for (int r = 0; r < 8; ++r) {
                int row = m0 + wm * 64 + mf * 16 + rsel + r;
                out[(size_t)row * O + ng] = acc[mf][nf][r] + bias;
            }
        }
    }
}

extern "C" void kernel_launch(void* const* d_in, const int* in_sizes, int n_in,
                              void* d_out, int out_size, void* d_ws, size_t ws_size,
                              hipStream_t stream) {
    const float* x       = (const float*)d_in[0];  // [B, 512]
    const float* coeff   = (const float*)d_in[1];  // [512, 512, 12]
    const float* base_w  = (const float*)d_in[2];  // [512, 512]
    const float* base_b  = (const float*)d_in[3];  // [512]
    const float* centers = (const float*)d_in[4];  // [12]
    float* out = (float*)d_out;

    const int I = 512, O = 512;
    const int B = in_sizes[0] / I;                 // 16384
    dim3 grid(O / TILE_N, B / TILE_M);             // (4, 128)
    rbfkan_wmma_kernel<<<grid, NTHREADS, 0, stream>>>(x, coeff, base_w, base_b,
                                                      centers, out);
    (void)d_ws; (void)ws_size; (void)n_in; (void)out_size;
}
